// ExpanderLinear_70179765616942
// MI455X (gfx1250) — matrix operand order repaired and checked by reading, not compiled, not run on hardware
//
#include <hip/hip_runtime.h>
#include <hip/hip_bf16.h>

typedef __attribute__((ext_vector_type(16))) _Float16 v16h;
typedef __attribute__((ext_vector_type(8)))  _Float16 v8h;
typedef __attribute__((ext_vector_type(4)))  _Float16 v4h;
typedef __attribute__((ext_vector_type(8)))  float    v8f;
typedef int v4i __attribute__((vector_size(16)));

#define INDIM  1024
#define OUTDIM 1024
#define NNZ    65536

// ---------------------------------------------------------------------------
// Async global->LDS copy (gfx1250 GLOBAL_LOAD_ASYNC_TO_LDS_B128, ASYNCcnt).
// Param types from clang's diagnostic: global-AS int4*, (LDS-AS int4*), imm, imm.
// Guarded: falls back to a synchronous copy if the builtins are absent.
// ---------------------------------------------------------------------------
#if defined(__HIP_DEVICE_COMPILE__) && \
    __has_builtin(__builtin_amdgcn_global_load_async_to_lds_b128) && \
    __has_builtin(__builtin_amdgcn_s_wait_asynccnt)
#define USE_ASYNC 1
#else
#define USE_ASYNC 0
#endif

#if USE_ASYNC
typedef __attribute__((address_space(1))) v4i* g128_t;   // global int4*
typedef __attribute__((address_space(3))) v4i* l128_t;   // LDS    int4*
#endif

__device__ __forceinline__ void copy16_g2l(const void* g, void* l) {
#if USE_ASYNC
  __builtin_amdgcn_global_load_async_to_lds_b128((g128_t)g, (l128_t)l, 0, 0);
#else
  *(float4*)l = *(const float4*)g;
#endif
}
__device__ __forceinline__ void async_wait_le4() {
#if USE_ASYNC
  __builtin_amdgcn_s_wait_asynccnt(4);
#endif
}
__device__ __forceinline__ void async_wait_le0() {
#if USE_ASYNC
  __builtin_amdgcn_s_wait_asynccnt(0);
#endif
}

// ---------------------------------------------------------------------------
// Kernel 1: zero the dense weight scratch (4 MB, L2 resident afterwards)
// ---------------------------------------------------------------------------
__global__ void ExpanderLinear_zero_w(float* __restrict__ wd) {
  int i = blockIdx.x * 256 + threadIdx.x;   // grid sized exactly to 1<<20
  wd[i] = 0.0f;
}

// ---------------------------------------------------------------------------
// Kernel 2: scatter nnz values into dense W (duplicates accumulate == SpMM)
// ---------------------------------------------------------------------------
__global__ void ExpanderLinear_scatter_w(const float* __restrict__ val,
                                         const int* __restrict__ ei,
                                         float* __restrict__ wd) {
  int e = blockIdx.x * 256 + threadIdx.x;
  if (e < NNZ) {
    int r = ei[e];          // row   (output dim)
    int c = ei[NNZ + e];    // col   (input dim)
    atomicAdd(&wd[(size_t)r * INDIM + c], val[e]);
  }
}

// ---------------------------------------------------------------------------
// Kernel 3: split W into f16 hi/lo, packed directly in the per-lane WMMA
// B-fragment layout:
//   tile = kt*64 + ntb  (16x16x32 B tile), 1KB per tile
//   packed[(tile*32 + lane)*16 + e] = W[ntb*16 + (lane&15)][kt*32 + (lane>>4)*16 + e]
// so one block's 8 consecutive n-tiles for a given kt are one contiguous 8KB.
// ---------------------------------------------------------------------------
__global__ void ExpanderLinear_split_w(const float* __restrict__ wd,
                                       _Float16* __restrict__ pbh,
                                       _Float16* __restrict__ pbl) {
  int i = blockIdx.x * 256 + threadIdx.x;   // packed index, grid = 1<<20 / 256
  int e    = i & 15;
  int lane = (i >> 4) & 31;
  int ntb  = (i >> 9) & 63;
  int kt   = i >> 15;
  int n = ntb * 16 + (lane & 15);
  int k = kt * 32 + (lane >> 4) * 16 + e;
  float w = wd[(size_t)n * INDIM + k];
  _Float16 h = (_Float16)w;
  pbh[i] = h;
  pbl[i] = (_Float16)(w - (float)h);
}

// ---------------------------------------------------------------------------
// Kernel 4: split-precision WMMA GEMM
//   block tile: 128 (M) x 128 (N), K-step 32, 256 threads = 8 wave32 waves
//   A: f32 loaded to regs 1 step ahead, converted to f16 hi/lo in LDS once
//   B: pre-packed fragments, async-copied to double-buffered LDS (ASYNCcnt)
//   per tile per K-step: 3x v_wmma_f32_16x16x32_f16 (hi*hi + lo*hi + hi*lo)
// ---------------------------------------------------------------------------
__global__ __launch_bounds__(256) void ExpanderLinear_gemm(
    const float*    __restrict__ x,
    const _Float16* __restrict__ pbh,
    const _Float16* __restrict__ pbl,
    const float*    __restrict__ bias,
    float*          __restrict__ out,
    int nrows)
{
  // A tile: 128 rows x 32 k, hi/lo f16; stride 40 halves = 80B (16B aligned,
  // bank bases m*20 mod 64 all distinct -> conflict-free b128 reads)
  __shared__ _Float16 sXh[128][40];
  __shared__ _Float16 sXl[128][40];
  // B tiles: [buf][hi/lo][tile][lane] -> exact fragment layout, 32B per lane
  __shared__ v16h sB[2][2][8][32];

  const int t    = threadIdx.x;
  const int lane = t & 31;
  const int wv   = t >> 5;
  const int wm   = wv & 3;       // 0..3 -> M position (32 rows each)
  const int wn   = wv >> 2;      // 0..1 -> N position (64 cols each)
  const int hx   = lane >> 4;    // lane half (ISA layout: K/M offset)
  const int lr   = lane & 15;    // row/col within 16x16 tile

  const int mbase = blockIdx.y * 128;
  const int nbase = blockIdx.x * 128;
  const int ntb0  = nbase >> 4;          // first 16-wide n-tile of this block

  v8f acc[8];
  {
    v8f z = {};
    #pragma unroll
    for (int i = 0; i < 8; ++i) acc[i] = z;
  }

  float4 xr[4];
  auto load_x = [&](int kt) {
    #pragma unroll
    for (int i = 0; i < 4; ++i) {
      int cc = t + 256 * i;               // 0..1023
      int r  = cc >> 3;                   // 0..127
      int c4 = (cc & 7) << 2;             // 0,4,..,28
      float4 pv = {0.0f, 0.0f, 0.0f, 0.0f};
      int grow = mbase + r;
      if (grow < nrows)
        pv = *(const float4*)(x + (size_t)grow * INDIM + kt * 32 + c4);
      xr[i] = pv;
    }
  };
  auto store_x = [&]() {
    #pragma unroll
    for (int i = 0; i < 4; ++i) {
      int cc = t + 256 * i;
      int r  = cc >> 3;
      int c4 = (cc & 7) << 2;
      float fv[4] = {xr[i].x, xr[i].y, xr[i].z, xr[i].w};
      v4h hv, lv;
      #pragma unroll
      for (int j = 0; j < 4; ++j) {
        float f = fv[j];
        _Float16 h = (_Float16)f;
        hv[j] = h;
        lv[j] = (_Float16)(f - (float)h);
      }
      *(v4h*)&sXh[r][c4] = hv;
      *(v4h*)&sXl[r][c4] = lv;
    }
  };
  auto copy_b = [&](int kt, int buf) {
    const char* gh = (const char*)pbh + ((size_t)kt * 64 + ntb0) * 1024;
    const char* gl = (const char*)pbl + ((size_t)kt * 64 + ntb0) * 1024;
    char* lh = (char*)&sB[buf][0][0][0];
    char* ll = (char*)&sB[buf][1][0][0];
    #pragma unroll
    for (int i = 0; i < 2; ++i) {
      int c = (t + 256 * i) << 4;         // 16B chunks over 8KB
      copy16_g2l(gh + c, lh + c);
      copy16_g2l(gl + c, ll + c);
    }
  };

  // ---- prologue ----
  load_x(0);
  copy_b(0, 0);

  for (int kt = 0; kt < 32; ++kt) {
    const int cur = kt & 1;
    if (kt) __syncthreads();              // WAR: everyone done reading sX
    store_x();                            // convert+stage A tile kt
    if (kt + 1 < 32) {
      load_x(kt + 1);                     // next x tile -> regs (in flight)
      copy_b(kt + 1, cur ^ 1);            // next B tile -> other LDS buffer
      async_wait_le4();                   // tile kt's 4 async b128 are done
    } else {
      async_wait_le0();
    }
    __syncthreads();                      // A stores + all waves' B(kt) ready

    // ---- A fragments: two contiguous 8-half runs per lane (ds_load_b128) ----
    v16h ah[2], al[2];
    const int base = 16 * hx;
    #pragma unroll
    for (int mt = 0; mt < 2; ++mt) {
      int m = 32 * wm + 16 * mt + lr;
      v8h h0 = *(const v8h*)&sXh[m][base];
      v8h h1 = *(const v8h*)&sXh[m][base + 8];
      v8h l0 = *(const v8h*)&sXl[m][base];
      v8h l1 = *(const v8h*)&sXl[m][base + 8];
      ah[mt] = __builtin_shufflevector(h0, h1, 0,1,2,3,4,5,6,7,8,9,10,11,12,13,14,15);
      al[mt] = __builtin_shufflevector(l0, l1, 0,1,2,3,4,5,6,7,8,9,10,11,12,13,14,15);
    }

    // ---- B fragments: direct 32B vector loads + 3-product accumulation ----
    #pragma unroll
    for (int nt = 0; nt < 4; ++nt) {
      v16h bh = sB[cur][0][4 * wn + nt][lane];
      v16h bl = sB[cur][1][4 * wn + nt][lane];
      #pragma unroll
      for (int mt = 0; mt < 2; ++mt) {
        v8f c = acc[mt * 4 + nt];
        c = __builtin_amdgcn_wmma_f32_16x16x32_f16(false, ah[mt], false, bh,
                                                   (short)0, c, false, false);
        c = __builtin_amdgcn_wmma_f32_16x16x32_f16(false, al[mt], false, bh,
                                                   (short)0, c, false, false);
        c = __builtin_amdgcn_wmma_f32_16x16x32_f16(false, ah[mt], false, bl,
                                                   (short)0, c, false, false);
        acc[mt * 4 + nt] = c;
      }
    }
  }

  // ---- epilogue: bias add + store (C/D layout: row M = r + 8*half) ----
  #pragma unroll
  for (int nt = 0; nt < 4; ++nt) {
    int col = nbase + 64 * wn + 16 * nt + lr;
    float bv = bias[col];
    #pragma unroll
    for (int mt = 0; mt < 2; ++mt) {
      #pragma unroll
      for (int r = 0; r < 8; ++r) {
        int row = mbase + 32 * wm + 16 * mt + 8 * hx + r;
        if (row < nrows)
          out[(size_t)row * OUTDIM + col] = acc[mt * 4 + nt][r] + bv;
      }
    }
  }
}

// ---------------------------------------------------------------------------
// Host launcher
// ---------------------------------------------------------------------------
extern "C" void kernel_launch(void* const* d_in, const int* in_sizes, int n_in,
                              void* d_out, int out_size, void* d_ws, size_t ws_size,
                              hipStream_t stream) {
  const float* x    = (const float*)d_in[0];   // [N, 1024] f32
  const float* val  = (const float*)d_in[1];   // [65536]   f32
  const float* bias = (const float*)d_in[2];   // [1024]    f32
  const int*   ei   = (const int*)d_in[3];     // [2, 65536] int32 (row, col)
  float* out = (float*)d_out;

  const int nrows = in_sizes[0] / INDIM;       // 50000

  // workspace: dense W f32 (4MB) | packed B hi f16 (2MB) | packed B lo (2MB)
  char* ws = (char*)d_ws;
  float*    wd  = (float*)ws;
  _Float16* pbh = (_Float16*)(ws + (4u << 20));
  _Float16* pbl = (_Float16*)(ws + (6u << 20));

  ExpanderLinear_zero_w<<<(1 << 20) / 256, 256, 0, stream>>>(wd);
  ExpanderLinear_scatter_w<<<NNZ / 256, 256, 0, stream>>>(val, ei, wd);
  ExpanderLinear_split_w<<<(1 << 20) / 256, 256, 0, stream>>>(wd, pbh, pbl);

  dim3 grid(OUTDIM / 128, (nrows + 127) / 128);   // N-fastest for L2 x reuse
  ExpanderLinear_gemm<<<grid, 256, 0, stream>>>(x, pbh, pbl, bias, out, nrows);
}
